// SledAttention_24060406792983
// MI455X (gfx1250) — matrix operand order repaired and checked by reading, not compiled, not run on hardware
//
#include <hip/hip_runtime.h>
#include <hip/hip_bf16.h>
#include <math.h>

typedef __attribute__((ext_vector_type(16))) _Float16 v16h;
typedef __attribute__((ext_vector_type(8)))  _Float16 v8h;
typedef __attribute__((ext_vector_type(8)))  float    v8f;
typedef __attribute__((ext_vector_type(4)))  unsigned int u32x4;
typedef __attribute__((ext_vector_type(8)))  int          i32x8;
typedef __attribute__((ext_vector_type(4)))  int          i32x4;

#define Bsz   2
#define Nseq  4096
#define Cdim  512
#define Hn    8
#define Dh    64
#define QKVW  1536
#define BN    8192

#if __has_builtin(__builtin_amdgcn_tensor_load_to_lds) && \
    __has_builtin(__builtin_amdgcn_s_wait_tensorcnt)
#define SLED_TDM 1
#else
#define SLED_TDM 0
#endif

// ---- WMMA helpers -----------------------------------------------------------

static __device__ inline v16h ld16(const _Float16* p0, const _Float16* p1) {
  v8h lo = *(const v8h*)p0;
  v8h hi = *(const v8h*)p1;
  v16h r;
#pragma unroll
  for (int i = 0; i < 8; ++i) { r[i] = lo[i]; r[i + 8] = hi[i]; }
  return r;
}

static __device__ inline v8f wmma(v16h a, v16h b, v8f c) {
  // D = A(16x32 f16) * B(32x16 f16) + C(16x16 f32)
  return __builtin_amdgcn_wmma_f32_16x16x32_f16(false, a, false, b, (short)0, c,
                                                false, false);
}

#if SLED_TDM
// D# group 0: count=1 | lds_addr | global_addr[56:0] | type=2 ("image")
static __device__ inline u32x4 tdm_g0(unsigned lds_off, unsigned long long ga) {
  u32x4 g;
  g[0] = 1u;                                   // count=1, user descriptor
  g[1] = lds_off;                              // LDS byte address
  g[2] = (unsigned)ga;                         // global_addr[31:0]
  g[3] = ((unsigned)(ga >> 32) & 0x01ffffffu) | (2u << 30);  // [56:32] | type=2
  return g;
}
#endif

// ---- Kernel 1: qkv = x @ w_qkv  (fp32 in -> f16 out) ------------------------

__global__ __launch_bounds__(256) void k_gemm_qkv(const float* __restrict__ x,
                                                  const float* __restrict__ w,
                                                  _Float16* __restrict__ qkv) {
  __shared__ __align__(16) _Float16 sA[128 * 40];   // A tile, padded stride 40
  __shared__ __align__(16) _Float16 sB[64 * 40];    // B^T tile
  const int m0 = blockIdx.y * 128;
  const int n0 = blockIdx.x * 64;
  const int t = threadIdx.x;
  const int wv = t >> 5, lane = t & 31, hf = lane >> 4, ln = lane & 15;

  v8f acc[4] = {};

  for (int k0 = 0; k0 < Cdim; k0 += 32) {
    if (k0 + 32 < Cdim)  // prefetch next A chunk -> global_prefetch_b8
      __builtin_prefetch(x + (size_t)(m0 + (t >> 1)) * Cdim + k0 + 32, 0, 1);
    // A: 128x32 fp32 -> f16 (coalesced float4 loads)
#pragma unroll
    for (int i = 0; i < 4; ++i) {
      int off = (t + 256 * i) * 4;                 // 4096 elements
      int r = off >> 5, kk = off & 31;
      float4 vx = *(const float4*)(x + (size_t)(m0 + r) * Cdim + k0 + kk);
      _Float16* d = &sA[r * 40 + kk];
      d[0] = (_Float16)vx.x; d[1] = (_Float16)vx.y;
      d[2] = (_Float16)vx.z; d[3] = (_Float16)vx.w;
    }
    // B^T: read w[k][n] coalesced, store transposed
#pragma unroll
    for (int i = 0; i < 2; ++i) {
      int off = (t + 256 * i) * 4;                 // 2048 elements
      int kk = off >> 6, nn = off & 63;
      float4 vw = *(const float4*)(w + (size_t)(k0 + kk) * QKVW + n0 + nn);
      sB[(nn + 0) * 40 + kk] = (_Float16)vw.x;
      sB[(nn + 1) * 40 + kk] = (_Float16)vw.y;
      sB[(nn + 2) * 40 + kk] = (_Float16)vw.z;
      sB[(nn + 3) * 40 + kk] = (_Float16)vw.w;
    }
    __syncthreads();

    const _Float16* ap = &sA[(16 * wv + ln) * 40];
    v16h a = ld16(ap + hf * 16, ap + hf * 16 + 8);
#pragma unroll
    for (int tt = 0; tt < 4; ++tt) {
      const _Float16* bp = &sB[(16 * tt + ln) * 40 + hf * 16];
      v16h b = ld16(bp, bp + 8);
      acc[tt] = wmma(a, b, acc[tt]);
    }
    __syncthreads();
  }

#pragma unroll
  for (int tt = 0; tt < 4; ++tt)
#pragma unroll
    for (int e = 0; e < 8; ++e) {
      int row = m0 + 16 * wv + e + 8 * hf;
      int col = n0 + 16 * tt + ln;
      qkv[(size_t)row * QKVW + col] = (_Float16)acc[tt][e];
    }
}

// ---- Kernel 2: RoPE + head scatter ------------------------------------------
// qh,kh: [B,H,N,64]  (q pre-scaled by 1/sqrt(d));  vt: [B,H,64,N] (transposed)

__global__ __launch_bounds__(256) void k_rope(const _Float16* __restrict__ qkv,
                                              const float* __restrict__ pos,
                                              _Float16* __restrict__ qh,
                                              _Float16* __restrict__ kh,
                                              _Float16* __restrict__ vt) {
  int gid = blockIdx.x * 256 + threadIdx.x;   // over BN * 512
  int row = gid >> 9;                         // b*N + n
  int c = gid & 511;
  int b = row >> 12, n = row & 4095;
  int h = c >> 6, d = c & 63;
  const _Float16* base = qkv + (size_t)row * QKVW;

  float q = (float)base[c];
  float k = (float)base[512 + c];
  float v = (float)base[1024 + c];
  int d2 = (d < 32) ? d + 32 : d - 32;
  float sgn = (d < 32) ? -1.f : 1.f;
  int c2 = h * 64 + d2;
  float q2 = sgn * (float)base[c2];
  float k2 = sgn * (float)base[512 + c2];
  float cs = __cosf(pos[n * 64 + d]);
  float sn = __sinf(pos[n * 64 + d]);

  size_t hoff = ((size_t)(b * Hn + h) * Nseq + n) * Dh + d;
  qh[hoff] = (_Float16)((q * cs + q2 * sn) * 0.125f);  // fold 1/sqrt(64)
  kh[hoff] = (_Float16)(k * cs + k2 * sn);
  vt[((size_t)(b * Hn + h) * Dh + d) * Nseq + n] = (_Float16)v;
}

// ---- Kernel 3: flash attention ----------------------------------------------
// grid = (N/128, B*H); 8 waves; each wave: 16 query rows, online softmax.
// K / V^T tiles staged by the Tensor Data Mover (TDM) when available; the
// D# pad fields (32 DWORDs interval, 4 DWORDs pad) produce the 72-half row
// stride directly in hardware.

__global__ __launch_bounds__(256) void k_attn(const _Float16* __restrict__ qh,
                                              const _Float16* __restrict__ kh,
                                              const _Float16* __restrict__ vt,
                                              _Float16* __restrict__ oh) {
  __shared__ __align__(16) _Float16 sK[64 * 72];       // K tile  [j][d]
  __shared__ __align__(16) _Float16 sV[64 * 72];       // V^T tile [d][j]
  __shared__ __align__(16) _Float16 sP[8 * 16 * 72];   // per-wave P tiles
  const int bh = blockIdx.y;
  const int b = bh >> 3, h = bh & 7;
  const int t = threadIdx.x;
  const int wv = t >> 5, lane = t & 31, hf = lane >> 4, ln = lane & 15;
  const int q0 = blockIdx.x * 128 + 16 * wv;

  // q A-fragments (K = 64 -> 2 fragments of K=32), loaded straight from global
  const _Float16* qp = qh + ((size_t)bh * Nseq + q0 + ln) * Dh;
  v16h qa[2];
#pragma unroll
  for (int c = 0; c < 2; ++c)
    qa[c] = ld16(qp + 32 * c + hf * 16, qp + 32 * c + hf * 16 + 8);

  float mrow[8], lrow[8];
#pragma unroll
  for (int e = 0; e < 8; ++e) { mrow[e] = -INFINITY; lrow[e] = 0.f; }
  v8f accO[4] = {};
  _Float16* pw = &sP[wv * 16 * 72];

#if SLED_TDM
  // D# group 1 descriptors (uniform across the loop):
  // dw0: data_size=2B | pad_enable | pad_interval=32dw | pad_amount=4dw
  const int G1CTL = (1 << 16) | (1 << 20) | (4 << 22) | (3 << 25);
  const i32x8 g1k = {G1CTL, 64 << 16, (int)(4096u << 16), 64 << 16, 64, 64, 0, 0};
  const i32x8 g1v = {G1CTL, (int)(4096u << 16), 64 << 16, 64 << 16, 64, 4096, 0, 0};
  const i32x4 z4 = {0, 0, 0, 0};
  const i32x8 z8 = {0, 0, 0, 0, 0, 0, 0, 0};
  const unsigned ldsK = (unsigned)(uintptr_t)&sK[0];
  const unsigned ldsV = (unsigned)(uintptr_t)&sV[0];
  const unsigned long long kg =
      (unsigned long long)(uintptr_t)(kh + (size_t)bh * Nseq * Dh);
  const unsigned long long vg =
      (unsigned long long)(uintptr_t)(vt + (size_t)bh * Dh * Nseq);
#endif

  for (int j0 = 0; j0 < Nseq; j0 += 64) {
#if SLED_TDM
    if (wv == 0) {        // K tile: 64 rows x 64 halves, row stride 64 halves
      __builtin_amdgcn_tensor_load_to_lds(
          tdm_g0(ldsK, kg + (unsigned long long)j0 * (Dh * 2)), g1k, z4, z4, z8,
          0);
    } else if (wv == 1) { // V^T tile: 64 rows x 64 halves, row stride 4096
      __builtin_amdgcn_tensor_load_to_lds(
          tdm_g0(ldsV, vg + (unsigned long long)j0 * 2), g1v, z4, z4, z8, 0);
    }
    if (wv < 2) __builtin_amdgcn_s_wait_tensorcnt(0);
    __syncthreads();
#else
    // cooperative load of K tile and V^T tile (v8h = 16B chunks)
#pragma unroll
    for (int i = 0; i < 2; ++i) {
      int off = (t + 256 * i) * 8;                // 4096 halves each
      int r = off >> 6, cc = off & 63;
      *(v8h*)&sK[r * 72 + cc] =
          *(const v8h*)(kh + ((size_t)bh * Nseq + j0 + r) * Dh + cc);
      *(v8h*)&sV[r * 72 + cc] =
          *(const v8h*)(vt + ((size_t)bh * Dh + r) * Nseq + j0 + cc);
    }
    if (j0 + 64 < Nseq)
      __builtin_prefetch(kh + ((size_t)bh * Nseq + j0 + 64 + (t >> 2)) * Dh, 0, 1);
    __syncthreads();
#endif

    // S = q . k^T  (16 x 64): B operand is k row-major (== (k^T)^T)
    v8f s[4] = {};
#pragma unroll
    for (int c = 0; c < 2; ++c)
#pragma unroll
      for (int tt = 0; tt < 4; ++tt) {
        const _Float16* bp = &sK[(16 * tt + ln) * 72 + 32 * c + hf * 16];
        v16h kb = ld16(bp, bp + 8);
        s[tt] = wmma(qa[c], kb, s[tt]);
      }

    // online softmax: row lives across the 16 lanes of this half
    float mt[8];
#pragma unroll
    for (int e = 0; e < 8; ++e) {
      float v0 = fmaxf(fmaxf(s[0][e], s[1][e]), fmaxf(s[2][e], s[3][e]));
#pragma unroll
      for (int o = 8; o >= 1; o >>= 1) v0 = fmaxf(v0, __shfl_xor(v0, o, 32));
      mt[e] = fmaxf(mrow[e], v0);
    }
    float alpha[8], rsum[8];
#pragma unroll
    for (int e = 0; e < 8; ++e) {
      alpha[e] = __expf(mrow[e] - mt[e]);
      mrow[e] = mt[e];
      rsum[e] = 0.f;
    }
#pragma unroll
    for (int tt = 0; tt < 4; ++tt)
#pragma unroll
      for (int e = 0; e < 8; ++e) {
        float p = __expf(s[tt][e] - mrow[e]);
        rsum[e] += p;
        pw[(e + 8 * hf) * 72 + 16 * tt + ln] = (_Float16)p;  // C->A via LDS
      }
#pragma unroll
    for (int e = 0; e < 8; ++e) {
      float r0 = rsum[e];
#pragma unroll
      for (int o = 8; o >= 1; o >>= 1) r0 += __shfl_xor(r0, o, 32);
      lrow[e] = lrow[e] * alpha[e] + r0;
#pragma unroll
      for (int tt = 0; tt < 4; ++tt) accO[tt][e] *= alpha[e];
    }
    __builtin_amdgcn_wave_barrier();  // wave-private LDS RAW ordering

    // O += P . V : A = P(16x64), B^T = V^T tile (row-major [d][j])
#pragma unroll
    for (int c = 0; c < 2; ++c) {
      v16h pa = ld16(pw + ln * 72 + 32 * c + hf * 16,
                     pw + ln * 72 + 32 * c + hf * 16 + 8);
#pragma unroll
      for (int td = 0; td < 4; ++td) {
        const _Float16* bp = &sV[(16 * td + ln) * 72 + 32 * c + hf * 16];
        v16h vb = ld16(bp, bp + 8);
        accO[td] = wmma(pa, vb, accO[td]);
      }
    }
    __syncthreads();
  }

  // normalize + write concat-head layout oh[b*N + q][h*64 + d]
#pragma unroll
  for (int td = 0; td < 4; ++td)
#pragma unroll
    for (int e = 0; e < 8; ++e) {
      int qr = q0 + e + 8 * hf;
      int d = 16 * td + ln;
      oh[((size_t)(b * Nseq + qr)) * Cdim + h * 64 + d] =
          (_Float16)(accO[td][e] / lrow[e]);
    }
}

// ---- Kernel 4: out = oh @ w_out + b_out  (f16 in -> fp32 out) ---------------

__global__ __launch_bounds__(256) void k_gemm_out(const _Float16* __restrict__ oh,
                                                  const float* __restrict__ w,
                                                  const float* __restrict__ bias,
                                                  float* __restrict__ out) {
  __shared__ __align__(16) _Float16 sA[128 * 40];
  __shared__ __align__(16) _Float16 sB[64 * 40];
  const int m0 = blockIdx.y * 128, n0 = blockIdx.x * 64;
  const int t = threadIdx.x;
  const int wv = t >> 5, lane = t & 31, hf = lane >> 4, ln = lane & 15;
  v8f acc[4] = {};

  for (int k0 = 0; k0 < Cdim; k0 += 32) {
    if (k0 + 32 < Cdim)
      __builtin_prefetch(oh + (size_t)(m0 + (t >> 1)) * Cdim + k0 + 32, 0, 1);
#pragma unroll
    for (int i = 0; i < 2; ++i) {
      int off = (t + 256 * i) * 8;               // 4096 halves
      int r = off >> 5, kk = off & 31;
      *(v8h*)&sA[r * 40 + kk] =
          *(const v8h*)(oh + (size_t)(m0 + r) * Cdim + k0 + kk);
    }
#pragma unroll
    for (int i = 0; i < 2; ++i) {
      int off = (t + 256 * i) * 4;
      int kk = off >> 6, nn = off & 63;
      float4 vw = *(const float4*)(w + (size_t)(k0 + kk) * Cdim + n0 + nn);
      sB[(nn + 0) * 40 + kk] = (_Float16)vw.x;
      sB[(nn + 1) * 40 + kk] = (_Float16)vw.y;
      sB[(nn + 2) * 40 + kk] = (_Float16)vw.z;
      sB[(nn + 3) * 40 + kk] = (_Float16)vw.w;
    }
    __syncthreads();

    const _Float16* ap = &sA[(16 * wv + ln) * 40];
    v16h a = ld16(ap + hf * 16, ap + hf * 16 + 8);
#pragma unroll
    for (int tt = 0; tt < 4; ++tt) {
      const _Float16* bp = &sB[(16 * tt + ln) * 40 + hf * 16];
      v16h bb = ld16(bp, bp + 8);
      acc[tt] = wmma(a, bb, acc[tt]);
    }
    __syncthreads();
  }

#pragma unroll
  for (int tt = 0; tt < 4; ++tt)
#pragma unroll
    for (int e = 0; e < 8; ++e) {
      int row = m0 + 16 * wv + e + 8 * hf;
      int col = n0 + 16 * tt + ln;
      out[(size_t)row * Cdim + col] = acc[tt][e] + bias[col];
    }
}

// ---- host-side launcher -----------------------------------------------------

extern "C" void kernel_launch(void* const* d_in, const int* in_sizes, int n_in,
                              void* d_out, int out_size, void* d_ws, size_t ws_size,
                              hipStream_t stream) {
  const float* x    = (const float*)d_in[0];
  const float* pos  = (const float*)d_in[1];
  const float* wqkv = (const float*)d_in[2];
  const float* wout = (const float*)d_in[3];
  const float* bout = (const float*)d_in[4];
  float* out = (float*)d_out;

  char* ws = (char*)d_ws;
  _Float16* qkv = (_Float16*)ws;                        // 8192*1536 f16 = 25.2MB
  _Float16* qh  = (_Float16*)(ws + 25165824);           // [B,H,N,64]   8MB
  _Float16* kh  = (_Float16*)(ws + 33554432);           // [B,H,N,64]   8MB
  _Float16* vt  = (_Float16*)(ws + 41943040);           // [B,H,64,N]   8MB
  _Float16* oh  = (_Float16*)ws;                        // reuse qkv region

  k_gemm_qkv<<<dim3(QKVW / 64, BN / 128), 256, 0, stream>>>(x, wqkv, qkv);
  k_rope<<<(BN * Cdim) / 256, 256, 0, stream>>>(qkv, pos, qh, kh, vt);
  k_attn<<<dim3(Nseq / 128, Bsz * Hn), 256, 0, stream>>>(qh, kh, vt, oh);
  k_gemm_out<<<dim3(Cdim / 64, BN / 128), 256, 0, stream>>>(oh, wout, bout, out);
}
